// AttentionLayer_55061480734844
// MI455X (gfx1250) — matrix operand order repaired and checked by reading, not compiled, not run on hardware
//
#include <hip/hip_runtime.h>
#include <hip/hip_bf16.h>
#include <math.h>

// ---------------------------------------------------------------------------
// Fused attention for MI455X (gfx1250, wave32, WMMA, TDM).
//   B=8, L=1024, C=1024, H=16, D=64
// Pipeline:
//   k0: fp32 -> bf16 convert (x, W_qkv)
//   k1: QKV GEMM: TDM (tensor_load_to_lds) double-buffered 128x32 tiles into
//       LDS, ds_load_b128 fragments, bf16 WMMA w/ fp32 accum; scatter q/k/v^T
//   k2: fused flash-attention per (b,h,16-row q tile), online softmax,
//       global_prefetch of next K/V tiles
// Workspace layout (72 MB total):
//   [ 0,16)MB x_bf16   [16,22)MB W_bf16
//   [24,40)MB q_bf16 [B,H,L,D]  [40,56)MB k_bf16 [B,H,L,D]  [56,72)MB vT [B,H,D,L]
// ---------------------------------------------------------------------------

typedef __bf16 bf16;
typedef __attribute__((ext_vector_type(16))) __bf16 v16bf;
typedef __attribute__((ext_vector_type(8)))  __bf16 v8bf;
typedef __attribute__((ext_vector_type(4)))  __bf16 v4bf;
typedef __attribute__((ext_vector_type(8)))  float  v8f;
typedef __attribute__((ext_vector_type(4)))  unsigned int u32x4;
typedef __attribute__((ext_vector_type(8)))  int i32x8;
typedef __attribute__((ext_vector_type(4)))  int i32x4;

#define NEG_FILL (-1e15f)
#define EPS      (0.01f)

union FragBF { v16bf v; v8bf h[2]; };

__device__ __forceinline__ v8f wmma_bf16(const FragBF& a, const FragBF& b, v8f c) {
  return __builtin_amdgcn_wmma_f32_16x16x32_bf16(
      false, a.v, false, b.v, (short)0, c, false, false);
}

// --- Tensor Data Mover: 2D bf16 tile (tile_w elems x tile_h rows) -> LDS ----
// D# per CDNA5 ISA. tensor_dim == tile_dim (no OOB),
// row stride in elements. Issued by one wave; tracked on TENSORcnt.
__device__ __forceinline__ void tdm_load_2d(unsigned int lds_off, const void* gptr,
                                            unsigned int tile_w, unsigned int tile_h,
                                            unsigned int row_stride_elems) {
  unsigned long long ga = (unsigned long long)gptr;
  u32x4 g0;
  g0[0] = 1u;                                   // count=1, user descriptor
  g0[1] = lds_off;                              // LDS byte address
  g0[2] = (unsigned int)ga;                     // global_addr[31:0]
  g0[3] = (unsigned int)((ga >> 32) & 0x1FFFFFFu) | (2u << 30);  // [56:32] | type=2
  i32x8 g1;
  g1[0] = (int)(1u << 16);                      // workgroup_mask=0, data_size=1 (2B)
  g1[1] = (int)((tile_w & 0xFFFFu) << 16);      // tensor_dim0[15:0] @ bits63:48
  g1[2] = (int)(((tile_w >> 16) & 0xFFFFu) | ((tile_h & 0xFFFFu) << 16)); // dim0hi|dim1lo
  g1[3] = (int)(((tile_h >> 16) & 0xFFFFu) | ((tile_w & 0xFFFFu) << 16)); // dim1hi|tile_dim0
  g1[4] = (int)(tile_h & 0xFFFFu);              // tile_dim1, tile_dim2=0
  g1[5] = (int)row_stride_elems;                // tensor_dim0_stride[31:0]
  g1[6] = 0;                                    // stride0 hi | dim1_stride lo
  g1[7] = 0;                                    // dim1_stride hi (unused, 2D)
  i32x4 z4 = {};
#if defined(__clang_major__) && (__clang_major__ >= 23)
  i32x8 z8 = {};
  __builtin_amdgcn_tensor_load_to_lds(g0, g1, z4, z4, z8, 0);
#else
  __builtin_amdgcn_tensor_load_to_lds(g0, g1, z4, z4, 0);
#endif
}

// ---------------------------------------------------------------- k0: convert
__global__ __launch_bounds__(256) void cvt_f32_bf16(const float* __restrict__ s,
                                                    bf16* __restrict__ d, int n) {
  int i = (blockIdx.x * blockDim.x + threadIdx.x) * 4;
  if (i + 3 < n) {
    float4 f = *(const float4*)(s + i);
    v4bf o;
    o[0] = (bf16)f.x; o[1] = (bf16)f.y; o[2] = (bf16)f.z; o[3] = (bf16)f.w;
    *(v4bf*)(d + i) = o;
  }
}

// ------------------------------------------------------------- k1: QKV GEMM
// Block tile 128x128, 8 waves (2M x 4N), wave tile 64x32. K-step 32.
// TDM double-buffers X[128x32] and W[128x32] tiles in LDS (4x/2x reuse).
__global__ __launch_bounds__(256) void qkv_gemm(const bf16* __restrict__ X,
                                                const bf16* __restrict__ W,
                                                bf16* __restrict__ qb,
                                                bf16* __restrict__ kb,
                                                bf16* __restrict__ vtb) {
  __shared__ bf16 sx[2][128 * 32];   // 2 x 8 KB
  __shared__ bf16 sw[2][128 * 32];   // 2 x 8 KB

  const int lane = threadIdx.x & 31;
  const int wave = threadIdx.x >> 5;
  const int n16  = lane & 15;
  const int g    = lane >> 4;

  const int m0blk = blockIdx.x * 128;
  const int n0blk = blockIdx.y * 128;
  const int wm = (wave & 1) * 64;
  const int wn = (wave >> 1) * 32;

  v8f acc[4][2] = {};

  if (wave == 0) {
    tdm_load_2d((unsigned int)(uintptr_t)&sx[0][0], X + (size_t)m0blk * 1024, 32, 128, 1024);
    tdm_load_2d((unsigned int)(uintptr_t)&sw[0][0], W + (size_t)n0blk * 1024, 32, 128, 1024);
  }

  for (int kk = 0; kk < 1024; kk += 32) {
    const int cur = (kk >> 5) & 1;
    const int nxt = cur ^ 1;
    if (wave == 0) __builtin_amdgcn_s_wait_tensorcnt(0);  // buf[cur] landed
    __syncthreads();                                      // publish; prev reads done
    if (wave == 0 && kk + 32 < 1024) {                    // overlap next DMA w/ math
      tdm_load_2d((unsigned int)(uintptr_t)&sx[nxt][0],
                  X + (size_t)m0blk * 1024 + kk + 32, 32, 128, 1024);
      tdm_load_2d((unsigned int)(uintptr_t)&sw[nxt][0],
                  W + (size_t)n0blk * 1024 + kk + 32, 32, 128, 1024);
    }

    const bf16* px = &sx[cur][0];
    const bf16* pw = &sw[cur][0];
    FragBF af[4], wf[2];
#pragma unroll
    for (int tm = 0; tm < 4; ++tm) {
      const bf16* row = px + (wm + tm * 16 + n16) * 32;
      af[tm].h[0] = *(const v8bf*)(row + 8 * g);
      af[tm].h[1] = *(const v8bf*)(row + 16 + 8 * g);
    }
#pragma unroll
    for (int tn = 0; tn < 2; ++tn) {
      const bf16* row = pw + (wn + tn * 16 + n16) * 32;
      wf[tn].h[0] = *(const v8bf*)(row + 8 * g);
      wf[tn].h[1] = *(const v8bf*)(row + 16 + 8 * g);
    }
#pragma unroll
    for (int tm = 0; tm < 4; ++tm)
#pragma unroll
      for (int tn = 0; tn < 2; ++tn)
        acc[tm][tn] = wmma_bf16(af[tm], wf[tn], acc[tm][tn]);
  }

  // Epilogue: scatter into q (scaled by D^-1/2 = 0.125), k, and v^T.
#pragma unroll
  for (int tm = 0; tm < 4; ++tm) {
#pragma unroll
    for (int tn = 0; tn < 2; ++tn) {
      const int col   = n0blk + wn + tn * 16 + n16;  // [0, 3072)
      const int three = col >> 10;
      const int h     = (col & 1023) >> 6;
      const int d     = col & 63;
#pragma unroll
      for (int r = 0; r < 8; ++r) {
        const int row = m0blk + wm + tm * 16 + r + 8 * g;
        const int b = row >> 10, l = row & 1023;
        const float val = acc[tm][tn][r];
        if (three == 0) {
          qb[(((size_t)(b * 16 + h)) * 1024 + l) * 64 + d] = (bf16)(val * 0.125f);
        } else if (three == 1) {
          kb[(((size_t)(b * 16 + h)) * 1024 + l) * 64 + d] = (bf16)val;
        } else {
          vtb[(((size_t)(b * 16 + h)) * 64 + d) * 1024 + l] = (bf16)val;
        }
      }
    }
  }
}

// --------------------------------------------------- k2: fused flash attention
__global__ __launch_bounds__(128) void attn_fused(const bf16* __restrict__ qb,
                                                  const bf16* __restrict__ kb,
                                                  const bf16* __restrict__ vtb,
                                                  const int* __restrict__ mask,
                                                  float* __restrict__ out) {
  __shared__ bf16 pls[4][16 * 32];

  const int lane = threadIdx.x & 31;
  const int wave = threadIdx.x >> 5;
  const int n16  = lane & 15;
  const int g    = lane >> 4;

  const int idx   = blockIdx.x;
  const int qblk  = idx & 15;
  const int h     = (idx >> 4) & 15;
  const int b     = idx >> 8;
  const int qbase = qblk * 64 + wave * 16;

  const bf16* qptr = qb  + (((size_t)(b * 16 + h)) * 1024 + qbase) * 64;
  const bf16* kptr = kb  + ((size_t)(b * 16 + h)) * 1024 * 64;
  const bf16* vptr = vtb + ((size_t)(b * 16 + h)) * 64 * 1024;
  const int*  mrow = mask + b * 1024;

  FragBF qf[2];
  {
    const bf16* row = qptr + (size_t)n16 * 64;
    qf[0].h[0] = *(const v8bf*)(row + 8 * g);
    qf[0].h[1] = *(const v8bf*)(row + 16 + 8 * g);
    qf[1].h[0] = *(const v8bf*)(row + 32 + 8 * g);
    qf[1].h[1] = *(const v8bf*)(row + 48 + 8 * g);
  }

  float mq[8];
#pragma unroll
  for (int r = 0; r < 8; ++r) mq[r] = (float)mrow[qbase + r + 8 * g];

  float mrun[8], ssum[8];
#pragma unroll
  for (int r = 0; r < 8; ++r) { mrun[r] = -INFINITY; ssum[r] = 0.f; }
  v8f o[4] = {};

  bf16* myp = &pls[wave][0];

  for (int j = 0; j < 1024; j += 32) {
    // prefetch next step's K/V rows into cache (global_prefetch_b8)
    if (j + 32 < 1024) {
      __builtin_prefetch(kptr + (size_t)(j + 32 + n16) * 64, 0, 0);
      __builtin_prefetch(kptr + (size_t)(j + 48 + n16) * 64, 0, 0);
      __builtin_prefetch(vptr + (size_t)n16 * 1024 + j + 32, 0, 0);
      __builtin_prefetch(vptr + (size_t)(16 + n16) * 1024 + j + 32, 0, 0);
    }

    v8f s[2] = {};
#pragma unroll
    for (int t = 0; t < 2; ++t) {
      FragBF bk0, bk1;
      const bf16* krow = kptr + (size_t)(j + 16 * t + n16) * 64;
      bk0.h[0] = *(const v8bf*)(krow + 8 * g);
      bk0.h[1] = *(const v8bf*)(krow + 16 + 8 * g);
      bk1.h[0] = *(const v8bf*)(krow + 32 + 8 * g);
      bk1.h[1] = *(const v8bf*)(krow + 48 + 8 * g);
      s[t] = wmma_bf16(qf[0], bk0, s[t]);
      s[t] = wmma_bf16(qf[1], bk1, s[t]);
    }

    const float mk0 = (float)mrow[j + n16];
    const float mk1 = (float)mrow[j + 16 + n16];

#pragma unroll
    for (int r = 0; r < 8; ++r) {
      float s0 = s[0][r], s1 = s[1][r];
      if (mq[r] * mk0 == 0.f) s0 = NEG_FILL;
      if (mq[r] * mk1 == 0.f) s1 = NEG_FILL;
      float mx = fmaxf(s0, s1);
#pragma unroll
      for (int w = 1; w < 16; w <<= 1) mx = fmaxf(mx, __shfl_xor(mx, w, 32));
      const float mnew = fmaxf(mrun[r], mx);
      const float p0 = __expf(s0 - mnew);
      const float p1 = __expf(s1 - mnew);
      float ps = p0 + p1;
#pragma unroll
      for (int w = 1; w < 16; w <<= 1) ps += __shfl_xor(ps, w, 32);
      const float resc = __expf(mrun[r] - mnew);
      ssum[r] = ssum[r] * resc + ps;
      mrun[r] = mnew;
#pragma unroll
      for (int dt = 0; dt < 4; ++dt) o[dt][r] *= resc;
      myp[(r + 8 * g) * 32 + n16]      = (bf16)p0;
      myp[(r + 8 * g) * 32 + 16 + n16] = (bf16)p1;
    }

    FragBF pf;
    pf.h[0] = *(const v8bf*)(myp + n16 * 32 + 8 * g);
    pf.h[1] = *(const v8bf*)(myp + n16 * 32 + 16 + 8 * g);

#pragma unroll
    for (int dt = 0; dt < 4; ++dt) {
      FragBF bv;
      const bf16* vrow = vptr + (size_t)(dt * 16 + n16) * 1024 + j;
      bv.h[0] = *(const v8bf*)(vrow + 8 * g);
      bv.h[1] = *(const v8bf*)(vrow + 16 + 8 * g);
      o[dt] = wmma_bf16(pf, bv, o[dt]);
    }
  }

#pragma unroll
  for (int dt = 0; dt < 4; ++dt) {
#pragma unroll
    for (int r = 0; r < 8; ++r) {
      const int row = qbase + r + 8 * g;
      out[((size_t)b * 1024 + row) * 1024 + h * 64 + dt * 16 + n16] =
          o[dt][r] / (ssum[r] + EPS);
    }
  }
}

// ---------------------------------------------------------------------------
extern "C" void kernel_launch(void* const* d_in, const int* in_sizes, int n_in,
                              void* d_out, int out_size, void* d_ws, size_t ws_size,
                              hipStream_t stream) {
  const float* x    = (const float*)d_in[0];
  const int*   mask = (const int*)d_in[1];
  const float* W    = (const float*)d_in[2];
  float*       out  = (float*)d_out;

  char* ws = (char*)d_ws;
  const size_t MB = 1024ull * 1024ull;
  bf16* xbf = (bf16*)(ws);
  bf16* wbf = (bf16*)(ws + 16 * MB);
  bf16* qb  = (bf16*)(ws + 24 * MB);
  bf16* kb  = (bf16*)(ws + 40 * MB);
  bf16* vt  = (bf16*)(ws + 56 * MB);

  const int nx = 8 * 1024 * 1024;
  const int nw = 3 * 1024 * 1024;

  cvt_f32_bf16<<<nx / 4 / 256, 256, 0, stream>>>(x, xbf, nx);
  cvt_f32_bf16<<<nw / 4 / 256, 256, 0, stream>>>(W, wbf, nw);

  dim3 gA(64, 24);
  qkv_gemm<<<gA, 256, 0, stream>>>(xbf, wbf, qb, kb, vt);

  attn_fused<<<2048, 128, 0, stream>>>(qb, kb, vt, mask, out);
  (void)in_sizes; (void)n_in; (void)out_size; (void)ws_size;
}
